// DeepSeekMoE_26843545600829
// MI455X (gfx1250) — matrix operand order repaired and checked by reading, not compiled, not run on hardware
//
#include <hip/hip_runtime.h>
#include <hip/hip_bf16.h>
#include <cstdint>

// ---------------------------------------------------------------------------
// DeepSeek MoE on MI455X (gfx1250): bf16 WMMA GEMMs with gathered dispatch.
// All GEMM tile staging (activations + weights) is done by the Tensor Data
// Mover into double-buffered LDS; waves only do ds_load + v_wmma.
// ---------------------------------------------------------------------------

#define D_DIM 2048
#define H_DIM 2048
#define N_TOK 8192          // B*S = 4*2048
#define NE    7
#define MATN  (2048 * 2048) // every weight matrix is 2048x2048
#define BK    64            // K-step per pipeline stage

typedef __bf16 bf16_t;
typedef __bf16 v16bf __attribute__((ext_vector_type(16)));
typedef float  v8f   __attribute__((ext_vector_type(8)));
typedef unsigned int u32x4 __attribute__((ext_vector_type(4)));
typedef int          i32x4 __attribute__((ext_vector_type(4)));
typedef int          i32x8 __attribute__((ext_vector_type(8)));

// ---------------------------------------------------------------------------
// TDM: async DMA of a 2D tile (tile_k contiguous bf16 x tile_n rows) into LDS.
// Descriptor per CDNA5 ISA ch.8 (group0: count/lds/global/type, group1: dims).
// ---------------------------------------------------------------------------
__device__ __forceinline__ void tdm_load_tile_2d(unsigned lds_off,
                                                 const bf16_t* gptr,
                                                 unsigned tile_k, unsigned tile_n,
                                                 unsigned long long stride_elems) {
    unsigned long long ga = (unsigned long long)(uintptr_t)gptr;
    u32x4 g0;
    g0[0] = 1u;                                    // count=1, user descriptor
    g0[1] = lds_off;                               // lds_addr (bytes)
    g0[2] = (unsigned)ga;                          // global_addr[31:0]
    g0[3] = (unsigned)(ga >> 32) | (2u << 30);     // global_addr[56:32] | type=2
    const unsigned td0 = 2048, td1 = 2048;         // tensor dims (OOB bounds)
    i32x8 g1;
    g1[0] = (int)(1u << 16);                                               // data_size=1 (2B)
    g1[1] = (int)((td0 & 0xFFFFu) << 16);                                  // tensor_dim0 lo
    g1[2] = (int)(((td0 >> 16) & 0xFFFFu) | ((td1 & 0xFFFFu) << 16));      // td0 hi | td1 lo
    g1[3] = (int)(((td1 >> 16) & 0xFFFFu) | ((tile_k & 0xFFFFu) << 16));   // td1 hi | tile_dim0
    g1[4] = (int)(tile_n & 0xFFFFu);                                       // tile_dim1 (tile_dim2=0)
    g1[5] = (int)(unsigned)(stride_elems & 0xFFFFFFFFull);                 // dim0_stride lo
    g1[6] = (int)(unsigned)((stride_elems >> 32) & 0xFFFFull);             // dim0_stride hi
    g1[7] = 0;
    i32x4 gz = {0, 0, 0, 0};
#if defined(__clang_major__) && (__clang_major__ >= 23)
    i32x8 gz8 = {0, 0, 0, 0, 0, 0, 0, 0};
    __builtin_amdgcn_tensor_load_to_lds(g0, g1, gz, gz, gz8, 0);
#else
    __builtin_amdgcn_tensor_load_to_lds(g0, g1, gz, gz, 0);
#endif
}

// A-operand fragment (16x32 bf16): lane m = L%16; lanes 0-15 hold K 0..7,16..23;
// lanes 16-31 hold K 8..15,24..31 -> two 16B chunks from the LDS row.
__device__ __forceinline__ v16bf load_a_frag(const bf16_t* row, int half) {
    union { uint4 q[2]; v16bf v; } r;
    r.q[0] = *(const uint4*)(row + half * 8);
    r.q[1] = *(const uint4*)(row + 16 + half * 8);
    return r.v;
}

// B-operand fragment (32x16 bf16): lane n = L%16 holds column n; lanes 0-15
// K=0..7,16..23, lanes 16-31 K=8..15,24..31. B staged K-contiguous per column in LDS.
__device__ __forceinline__ v16bf load_b_frag(const bf16_t* col, int half) {
    union { uint4 q[2]; v16bf v; } r;
    r.q[0] = *(const uint4*)(col + half * 8);
    r.q[1] = *(const uint4*)(col + 16 + half * 8);
    return r.v;
}

// ---------------------------------------------------------------------------
// x fp32 -> bf16
// ---------------------------------------------------------------------------
__global__ __launch_bounds__(256) void convert_x_bf16(const float* __restrict__ x,
                                                      bf16_t* __restrict__ xb) {
    int i = blockIdx.x * 256 + threadIdx.x;          // one float4 per thread
    float4 v = ((const float4*)x)[i];
    union { bf16_t b[4]; uint2 u; } p;
    p.b[0] = (bf16_t)v.x; p.b[1] = (bf16_t)v.y; p.b[2] = (bf16_t)v.z; p.b[3] = (bf16_t)v.w;
    *(uint2*)(xb + 4 * (size_t)i) = p.u;
}

// ---------------------------------------------------------------------------
// 2048x2048 transpose + fp32->bf16 for all 24 weight matrices.
// z: 0..6 W1[e], 7..13 Wg[e], 14..20 W2[e], 21 Ws1, 22 Wsg, 23 Ws2.
// Output WT[z] is "N-major": row n holds the K-contiguous reduction axis.
// ---------------------------------------------------------------------------
__global__ void transpose_to_bf16(const float* __restrict__ W1, const float* __restrict__ Wg,
                                  const float* __restrict__ W2, const float* __restrict__ Ws1,
                                  const float* __restrict__ Wsg, const float* __restrict__ Ws2,
                                  bf16_t* __restrict__ WT) {
    __shared__ float tile[32][33];
    int z = blockIdx.z;
    const float* src;
    if      (z < 7)  src = W1 + (size_t)z * MATN;
    else if (z < 14) src = Wg + (size_t)(z - 7) * MATN;
    else if (z < 21) src = W2 + (size_t)(z - 14) * MATN;
    else if (z == 21) src = Ws1;
    else if (z == 22) src = Wsg;
    else              src = Ws2;
    bf16_t* dst = WT + (size_t)z * MATN;
    int tx = threadIdx.x, ty = threadIdx.y;
    int x0 = blockIdx.x * 32, y0 = blockIdx.y * 32;
#pragma unroll
    for (int j = 0; j < 32; j += 8)
        tile[ty + j][tx] = src[(size_t)(y0 + ty + j) * 2048 + x0 + tx];
    __syncthreads();
#pragma unroll
    for (int j = 0; j < 32; j += 8)
        dst[(size_t)(x0 + ty + j) * 2048 + y0 + tx] = (bf16_t)tile[tx][ty + j];
}

// ---------------------------------------------------------------------------
// Gating: one wave per token. logits -> softmax -> top-3 -> normalized combine
// weights (dense [tok][7], zero if unselected) + gathered per-expert lists.
// ---------------------------------------------------------------------------
__global__ __launch_bounds__(256) void gate_topk(const float* __restrict__ x,
                                                 const float* __restrict__ gw,
                                                 const float* __restrict__ gb,
                                                 const float* __restrict__ biases,
                                                 float* __restrict__ wcomb,
                                                 int* __restrict__ lists,
                                                 int* __restrict__ cnt) {
    int wv = threadIdx.x >> 5, lane = threadIdx.x & 31;
    int tok = blockIdx.x * 8 + wv;
    float acc[NE];
#pragma unroll
    for (int e = 0; e < NE; ++e) acc[e] = 0.f;
    const float* xr = x + (size_t)tok * D_DIM;
    for (int d = lane; d < D_DIM; d += 32) {
        float xv = xr[d];
#pragma unroll
        for (int e = 0; e < NE; ++e) acc[e] += xv * gw[d * NE + e];
    }
#pragma unroll
    for (int off = 16; off > 0; off >>= 1) {
#pragma unroll
        for (int e = 0; e < NE; ++e) acc[e] += __shfl_xor(acc[e], off, 32);
    }
    if (lane == 0) {
        float logit[NE], p[NE], bp[NE];
        float m = -1e30f;
#pragma unroll
        for (int e = 0; e < NE; ++e) { logit[e] = acc[e] + gb[e]; m = fmaxf(m, logit[e]); }
        float s = 0.f;
#pragma unroll
        for (int e = 0; e < NE; ++e) { p[e] = __expf(logit[e] - m); s += p[e]; }
        float inv = 1.f / s;
#pragma unroll
        for (int e = 0; e < NE; ++e) { p[e] *= inv; bp[e] = p[e] + biases[e]; }
        bool sel[NE];
#pragma unroll
        for (int e = 0; e < NE; ++e) sel[e] = false;
        float sum3 = 0.f;
        for (int t = 0; t < 3; ++t) {
            int best = 0; float bv = -1e30f;
#pragma unroll
            for (int e = 0; e < NE; ++e)
                if (!sel[e] && bp[e] > bv) { bv = bp[e]; best = e; }
            sel[best] = true; sum3 += p[best];
        }
        float isum = 1.f / sum3;
#pragma unroll
        for (int e = 0; e < NE; ++e) {
            wcomb[(size_t)tok * NE + e] = sel[e] ? p[e] * isum : 0.f;
            if (sel[e]) {
                int pos = atomicAdd(&cnt[e], 1);
                lists[(size_t)e * N_TOK + pos] = tok;
            }
        }
    }
}

// ---------------------------------------------------------------------------
// Row gather: xg[i,:] = xb[list[i],:], i < count. One block per row
// (256 threads x uint4 = 4KB = one 2048-elem bf16 row).
// ---------------------------------------------------------------------------
__global__ __launch_bounds__(256) void gather_rows(const bf16_t* __restrict__ xb,
                                                   bf16_t* __restrict__ xg,
                                                   const int* __restrict__ list,
                                                   const int* __restrict__ cntp) {
    int i = blockIdx.x;
    if (i >= cntp[0]) return;
    int tok = list[i];
    const uint4* s = (const uint4*)(xb + (size_t)tok * D_DIM);
    uint4* d = (uint4*)(xg + (size_t)i * D_DIM);
    d[threadIdx.x] = s[threadIdx.x];
}

// ---------------------------------------------------------------------------
// GEMM1 (fused SwiGLU): h[i,:] = silu(xA[i]@W1) * (xA[i]@Wg), bf16 out.
// xA rows are compacted (pre-gathered), so A tiles are contiguous: A, W1, Wg
// tiles are ALL staged by TDM into double-buffered LDS; waves only do
// ds_load + v_wmma. One barrier per K-step; stage k+1 DMA overlaps compute k.
// Block tile 128(M) x 64(N), K-step 64; 8 waves = 4(M) x 2(N), 2x2 subtiles.
// ---------------------------------------------------------------------------
__global__ __launch_bounds__(256) void gemm1_swiglu(const bf16_t* __restrict__ xA,
                                                    const bf16_t* __restrict__ W1t,
                                                    const bf16_t* __restrict__ Wgt,
                                                    bf16_t* __restrict__ hbuf,
                                                    const int* __restrict__ cntp) {
    const int count = cntp ? cntp[0] : N_TOK;
    if ((int)blockIdx.y * 128 >= count) return;

    __shared__ __align__(16) bf16_t As[2][128 * BK];
    __shared__ __align__(16) bf16_t B1s[2][64 * BK];
    __shared__ __align__(16) bf16_t Bgs[2][64 * BK];

    const int tid = threadIdx.x;
    const int lane = tid & 31, wv = tid >> 5;
    const int waveM = wv & 3, waveN = wv >> 2;
    const int ln = lane & 15, half = lane >> 4;

    const bf16_t* abase  = xA  + (size_t)(blockIdx.y * 128) * D_DIM;
    const bf16_t* b1base = W1t + (size_t)(blockIdx.x * 64) * D_DIM;
    const bf16_t* bgbase = Wgt + (size_t)(blockIdx.x * 64) * D_DIM;
    unsigned alds[2]  = { (unsigned)(uintptr_t)&As[0][0],  (unsigned)(uintptr_t)&As[1][0] };
    unsigned b1lds[2] = { (unsigned)(uintptr_t)&B1s[0][0], (unsigned)(uintptr_t)&B1s[1][0] };
    unsigned bglds[2] = { (unsigned)(uintptr_t)&Bgs[0][0], (unsigned)(uintptr_t)&Bgs[1][0] };

    const v8f vzero = {0.f, 0.f, 0.f, 0.f, 0.f, 0.f, 0.f, 0.f};
    v8f acc1[2][2], accg[2][2];
#pragma unroll
    for (int sm = 0; sm < 2; ++sm)
#pragma unroll
        for (int sn = 0; sn < 2; ++sn) { acc1[sm][sn] = vzero; accg[sm][sn] = vzero; }

    // prologue: DMA stage 0
    if (wv == 0) {
        tdm_load_tile_2d(alds[0],  abase,  BK, 128, (unsigned long long)D_DIM);
        tdm_load_tile_2d(b1lds[0], b1base, BK, 64,  (unsigned long long)D_DIM);
        tdm_load_tile_2d(bglds[0], bgbase, BK, 64,  (unsigned long long)D_DIM);
    }

    const int KS = D_DIM / BK;
    for (int kk = 0; kk < KS; ++kk) {
        const int cur = kk & 1, nxt = cur ^ 1;
        if (wv == 0) __builtin_amdgcn_s_wait_tensorcnt(0);  // stage-cur DMAs done
        __syncthreads();  // stage-cur LDS ready; stage-nxt buffers free
        if (kk + 1 < KS && wv == 0) {   // stage-nxt DMA overlaps compute below
            const int k1 = (kk + 1) * BK;
            tdm_load_tile_2d(alds[nxt],  abase + k1,  BK, 128, (unsigned long long)D_DIM);
            tdm_load_tile_2d(b1lds[nxt], b1base + k1, BK, 64,  (unsigned long long)D_DIM);
            tdm_load_tile_2d(bglds[nxt], bgbase + k1, BK, 64,  (unsigned long long)D_DIM);
        }
        // compute stage cur: 2 sub-K x 2x2 subtiles x 2 matrices = 16 WMMA
#pragma unroll
        for (int ks = 0; ks < BK / 32; ++ks) {
            v16bf a[2], b1[2], bg[2];
#pragma unroll
            for (int sm = 0; sm < 2; ++sm)
                a[sm] = load_a_frag(&As[cur][(waveM * 32 + sm * 16 + ln) * BK + ks * 32], half);
#pragma unroll
            for (int sn = 0; sn < 2; ++sn) {
                b1[sn] = load_b_frag(&B1s[cur][(waveN * 32 + sn * 16 + ln) * BK + ks * 32], half);
                bg[sn] = load_b_frag(&Bgs[cur][(waveN * 32 + sn * 16 + ln) * BK + ks * 32], half);
            }
#pragma unroll
            for (int sm = 0; sm < 2; ++sm)
#pragma unroll
                for (int sn = 0; sn < 2; ++sn) {
                    acc1[sm][sn] = __builtin_amdgcn_wmma_f32_16x16x32_bf16(
                        false, a[sm], false, b1[sn], (short)0, acc1[sm][sn], false, false);
                    accg[sm][sn] = __builtin_amdgcn_wmma_f32_16x16x32_bf16(
                        false, a[sm], false, bg[sn], (short)0, accg[sm][sn], false, false);
                }
        }
    }

    // epilogue: silu(u) * g -> bf16 h (compacted token order)
#pragma unroll
    for (int sm = 0; sm < 2; ++sm)
#pragma unroll
        for (int sn = 0; sn < 2; ++sn)
#pragma unroll
            for (int j = 0; j < 8; ++j) {
                float u = acc1[sm][sn][j];
                float g = accg[sm][sn][j];
                float hv = __fdividef(u, 1.f + __expf(-u)) * g;
                int i = blockIdx.y * 128 + waveM * 32 + sm * 16 + half * 8 + j;
                int col = blockIdx.x * 64 + waveN * 32 + sn * 16 + ln;
                if (i < count) hbuf[(size_t)i * H_DIM + col] = (bf16_t)hv;
            }
}

// ---------------------------------------------------------------------------
// GEMM2: out[tok(i),:] (+)= w_e(tok) * (h[i,:] @ W2). h is compacted, so A
// and B tiles both stage via TDM. Scatter epilogue applies combine weight;
// accumulate=0 initializes out (shared expert covers every element).
// ---------------------------------------------------------------------------
__global__ __launch_bounds__(256) void gemm2_combine(const bf16_t* __restrict__ hbuf,
                                                     const bf16_t* __restrict__ W2t,
                                                     float* __restrict__ out,
                                                     const int* __restrict__ list,
                                                     const int* __restrict__ cntp,
                                                     const float* __restrict__ wcomb,
                                                     int expert, int accumulate) {
    const int count = cntp ? cntp[0] : N_TOK;
    if ((int)blockIdx.y * 128 >= count) return;

    __shared__ __align__(16) bf16_t As[2][128 * BK];
    __shared__ __align__(16) bf16_t Bs[2][64 * BK];

    const int tid = threadIdx.x;
    const int lane = tid & 31, wv = tid >> 5;
    const int waveM = wv & 3, waveN = wv >> 2;
    const int ln = lane & 15, half = lane >> 4;

    const bf16_t* abase = hbuf + (size_t)(blockIdx.y * 128) * H_DIM;
    const bf16_t* bbase = W2t + (size_t)(blockIdx.x * 64) * H_DIM;
    unsigned alds[2] = { (unsigned)(uintptr_t)&As[0][0], (unsigned)(uintptr_t)&As[1][0] };
    unsigned blds[2] = { (unsigned)(uintptr_t)&Bs[0][0], (unsigned)(uintptr_t)&Bs[1][0] };

    const v8f vzero = {0.f, 0.f, 0.f, 0.f, 0.f, 0.f, 0.f, 0.f};
    v8f acc[2][2];
#pragma unroll
    for (int sm = 0; sm < 2; ++sm)
#pragma unroll
        for (int sn = 0; sn < 2; ++sn) acc[sm][sn] = vzero;

    if (wv == 0) {
        tdm_load_tile_2d(alds[0], abase, BK, 128, (unsigned long long)H_DIM);
        tdm_load_tile_2d(blds[0], bbase, BK, 64,  (unsigned long long)H_DIM);
    }

    const int KS = H_DIM / BK;
    for (int kk = 0; kk < KS; ++kk) {
        const int cur = kk & 1, nxt = cur ^ 1;
        if (wv == 0) __builtin_amdgcn_s_wait_tensorcnt(0);
        __syncthreads();
        if (kk + 1 < KS && wv == 0) {
            const int k1 = (kk + 1) * BK;
            tdm_load_tile_2d(alds[nxt], abase + k1, BK, 128, (unsigned long long)H_DIM);
            tdm_load_tile_2d(blds[nxt], bbase + k1, BK, 64,  (unsigned long long)H_DIM);
        }
#pragma unroll
        for (int ks = 0; ks < BK / 32; ++ks) {
            v16bf a[2], b[2];
#pragma unroll
            for (int sm = 0; sm < 2; ++sm)
                a[sm] = load_a_frag(&As[cur][(waveM * 32 + sm * 16 + ln) * BK + ks * 32], half);
#pragma unroll
            for (int sn = 0; sn < 2; ++sn)
                b[sn] = load_b_frag(&Bs[cur][(waveN * 32 + sn * 16 + ln) * BK + ks * 32], half);
#pragma unroll
            for (int sm = 0; sm < 2; ++sm)
#pragma unroll
                for (int sn = 0; sn < 2; ++sn)
                    acc[sm][sn] = __builtin_amdgcn_wmma_f32_16x16x32_bf16(
                        false, a[sm], false, b[sn], (short)0, acc[sm][sn], false, false);
        }
    }

#pragma unroll
    for (int sm = 0; sm < 2; ++sm)
#pragma unroll
        for (int sn = 0; sn < 2; ++sn)
#pragma unroll
            for (int j = 0; j < 8; ++j) {
                int i = blockIdx.y * 128 + waveM * 32 + sm * 16 + half * 8 + j;
                if (i >= count) continue;
                int col = blockIdx.x * 64 + waveN * 32 + sn * 16 + ln;
                int tok = list ? list[i] : i;
                float w = wcomb ? wcomb[(size_t)tok * NE + expert] : 1.f;
                size_t o = (size_t)tok * D_DIM + col;
                float v = w * acc[sm][sn][j];
                if (accumulate) out[o] += v;   // experts serialized on stream
                else            out[o] = v;    // shared expert initializes out
            }
}

// ---------------------------------------------------------------------------
// Host launcher (graph-capture safe: kernels + hipMemsetAsync only).
// ---------------------------------------------------------------------------
extern "C" void kernel_launch(void* const* d_in, const int* in_sizes, int n_in,
                              void* d_out, int out_size, void* d_ws, size_t ws_size,
                              hipStream_t stream) {
    const float* x   = (const float*)d_in[0];
    const float* W1  = (const float*)d_in[1];
    const float* Wg  = (const float*)d_in[2];
    const float* W2  = (const float*)d_in[3];
    const float* Ws1 = (const float*)d_in[4];
    const float* Wsg = (const float*)d_in[5];
    const float* Ws2 = (const float*)d_in[6];
    const float* gw  = (const float*)d_in[7];
    const float* gb  = (const float*)d_in[8];
    const float* bia = (const float*)d_in[9];
    float* out = (float*)d_out;

    char* ws = (char*)d_ws;
    size_t off = 0;
    bf16_t* WT = (bf16_t*)(ws + off); off += (size_t)24 * MATN * sizeof(bf16_t);
    bf16_t* xb = (bf16_t*)(ws + off); off += (size_t)N_TOK * D_DIM * sizeof(bf16_t);
    bf16_t* xg = (bf16_t*)(ws + off); off += (size_t)N_TOK * D_DIM * sizeof(bf16_t);
    bf16_t* hb = (bf16_t*)(ws + off); off += (size_t)N_TOK * H_DIM * sizeof(bf16_t);
    float* wcomb = (float*)(ws + off); off += (size_t)N_TOK * NE * sizeof(float);
    int* lists = (int*)(ws + off); off += (size_t)NE * N_TOK * sizeof(int);
    int* cnt = (int*)(ws + off); off += 64;

    hipMemsetAsync(cnt, 0, NE * sizeof(int), stream);

    convert_x_bf16<<<dim3((N_TOK * D_DIM) / 4 / 256), dim3(256), 0, stream>>>(x, xb);
    transpose_to_bf16<<<dim3(64, 64, 24), dim3(32, 8), 0, stream>>>(W1, Wg, W2, Ws1, Wsg, Ws2, WT);
    gate_topk<<<dim3(N_TOK / 8), dim3(256), 0, stream>>>(x, gw, gb, bia, wcomb, lists, cnt);

    const dim3 ggrid(H_DIM / 64, N_TOK / 128);
    // shared expert: all tokens in natural order, weight 1, initializes out
    gemm1_swiglu<<<ggrid, dim3(256), 0, stream>>>(xb, WT + (size_t)21 * MATN,
                                                  WT + (size_t)22 * MATN, hb, nullptr);
    gemm2_combine<<<ggrid, dim3(256), 0, stream>>>(hb, WT + (size_t)23 * MATN, out,
                                                   nullptr, nullptr, nullptr, 0, 0);
    // routed experts: compact rows, then contiguous-tile GEMMs, accumulate
    for (int e = 0; e < NE; ++e) {
        gather_rows<<<dim3(N_TOK), dim3(256), 0, stream>>>(xb, xg,
                                                           lists + (size_t)e * N_TOK, cnt + e);
        gemm1_swiglu<<<ggrid, dim3(256), 0, stream>>>(xg, WT + (size_t)e * MATN,
                                                      WT + (size_t)(7 + e) * MATN, hb, cnt + e);
        gemm2_combine<<<ggrid, dim3(256), 0, stream>>>(hb, WT + (size_t)(14 + e) * MATN, out,
                                                       lists + (size_t)e * N_TOK, cnt + e,
                                                       wcomb, e, 1);
    }
}
